// TransformerBlock_56753697849917
// MI455X (gfx1250) — compile-verified, matrix operand order checked by reading, not run on hardware
//
#include <hip/hip_runtime.h>
#include <hip/hip_bf16.h>
#include <math.h>

// Problem dims (fixed by reference)
#define BB    2
#define SS    2048
#define DDIM  2048
#define HH    32
#define HDIM  128
#define PP    4096          // H*HD
#define KCONV 4
#define FDIM  8192
#define MM    (BB*SS)       // 4096 tokens
#define EPS   1e-6f

typedef __attribute__((ext_vector_type(16))) __bf16          v16bf;
typedef __attribute__((ext_vector_type(16))) unsigned short  v16u;
typedef __attribute__((ext_vector_type(8)))  float           v8f;

__device__ __forceinline__ unsigned short f2bf(float x) {
    union { float f; unsigned u; } c; c.f = x;
    unsigned u = c.u;
    u += 0x7fffu + ((u >> 16) & 1u);      // round-to-nearest-even
    return (unsigned short)(u >> 16);
}
__device__ __forceinline__ float sigmoidf_(float x) { return 1.0f / (1.0f + __expf(-x)); }
__device__ __forceinline__ float siluf_(float x)    { return x * sigmoidf_(x); }

// ---------------------------------------------------------------------------
// f32 -> bf16 conversion (4 elements/thread)
// ---------------------------------------------------------------------------
__global__ __launch_bounds__(256) void f32_to_bf16_kernel(
    const float* __restrict__ in, unsigned short* __restrict__ out, int n4)
{
    const int i = blockIdx.x * 256 + threadIdx.x;
    if (i >= n4) return;
    const float4 v = ((const float4*)in)[i];
    ((ushort4*)out)[i] = make_ushort4(f2bf(v.x), f2bf(v.y), f2bf(v.z), f2bf(v.w));
}

// ---------------------------------------------------------------------------
// WMMA GEMM on pre-converted bf16 operands:
//   C[M,N] = epilogue(A[M,K] * B[K,N])
// A tile (128x32 bf16, contiguous rows) staged with GLOBAL_LOAD_ASYNC_TO_LDS_B128
// (ASYNCcnt path); B tile staged synchronously with transpose into [n][k].
// Block tile 128x128, K-chunk 32, 8 waves: each wave 64x32 via 4x2
// v_wmma_f32_16x16x32_bf16 accumulators.
// mode: 0=C=AB ; 1=C=aux+AB ; 2=C=sigmoid(AB) ; 3=C=silu(AB)*aux
// obf : 1 -> store C as bf16 (ushort), 0 -> store f32
// Requires: M%128==0, K%32==0, N%8==0.
// ---------------------------------------------------------------------------
#define BM 128
#define BN 128
#define BK 32
#define LDPB (BK + 8)    // padded LDS row stride for B (ushort units)

__global__ __launch_bounds__(256) void gemm_bf16_wmma(
    const unsigned short* __restrict__ A, const unsigned short* __restrict__ Bw,
    void* __restrict__ Cv, const float* __restrict__ aux,
    int M, int N, int K, int mode, int obf)
{
    __shared__ unsigned short As[BM * BK];    // [m][k], rows of 64B, contiguous (async dest)
    __shared__ unsigned short Bs[BN][LDPB];   // [n][k] transposed, padded

    const int tid  = threadIdx.x;
    const int lane = tid & 31;
    const int wave = tid >> 5;
    const int wm   = wave >> 2;       // 0..1 : wave row (64 rows)
    const int wn   = wave & 3;        // 0..3 : wave col (32 cols)
    const int half = lane >> 4;       // 0/1
    const int l16  = lane & 15;

    const int rowBase = blockIdx.y * BM;
    const int colBase = blockIdx.x * BN;

    const unsigned asbase = (unsigned)(size_t)(&As[0]);   // low 32 bits = LDS byte offset

    v8f acc[4][2];
    #pragma unroll
    for (int m = 0; m < 4; ++m)
        #pragma unroll
        for (int n = 0; n < 2; ++n)
            #pragma unroll
            for (int e = 0; e < 8; ++e) acc[m][n][e] = 0.0f;

    const int nK = K / BK;
    for (int kt = 0; kt < nK; ++kt) {
        const int kBase = kt * BK;

        // ---- A tile: async global->LDS copy, 512 x 16B chunks, 2 per thread
        #pragma unroll
        for (int i = 0; i < 2; ++i) {
            const int c  = tid + i * 256;       // chunk 0..511
            const int r  = c >> 2;              // row 0..127
            const int q4 = c & 3;               // 16B quarter within 64B row
            const unsigned short* gp = &A[(size_t)(rowBase + r) * K + kBase + q4 * 8];
            const unsigned lds = asbase + (unsigned)(c * 16);
            asm volatile("global_load_async_to_lds_b128 %0, %1, off"
                         :: "v"(lds), "v"(gp) : "memory");
        }

        // ---- B tile: sync staged with transpose (bf16 global), 8 elems/iter
        #pragma unroll
        for (int i = 0; i < 2; ++i) {
            const int lin = tid + i * 256;      // 8-elem group 0..511
            const int kr  = lin >> 4;           // k row 0..31
            const int g8  = lin & 15;           // 8-col group within row
            const int col = colBase + g8 * 8;
            unsigned short t[8] = {0, 0, 0, 0, 0, 0, 0, 0};
            if (col + 8 <= N) {
                const uint4 raw = *(const uint4*)(&Bw[(size_t)(kBase + kr) * N + col]);
                t[0] = (unsigned short)(raw.x & 0xffffu); t[1] = (unsigned short)(raw.x >> 16);
                t[2] = (unsigned short)(raw.y & 0xffffu); t[3] = (unsigned short)(raw.y >> 16);
                t[4] = (unsigned short)(raw.z & 0xffffu); t[5] = (unsigned short)(raw.z >> 16);
                t[6] = (unsigned short)(raw.w & 0xffffu); t[7] = (unsigned short)(raw.w >> 16);
            }
            #pragma unroll
            for (int j = 0; j < 8; ++j)
                Bs[g8 * 8 + j][kr] = t[j];
        }

        // prefetch next B chunk into cache
        if (kt + 1 < nK)
            __builtin_prefetch(&Bw[(size_t)(kBase + BK + (tid >> 3)) * N + colBase], 0, 1);

        // complete this wave's async LDS writes, then workgroup barrier
        asm volatile("s_wait_asynccnt 0x0" ::: "memory");
        __syncthreads();

        // ---- fragment loads (A 16x32 bf16 layout: lanes0-15 carry M, half picks K group)
        v16bf afr[4], bfr[2];
        #pragma unroll
        for (int m = 0; m < 4; ++m) {
            union { v16u v; unsigned short s[16]; } u;
            const int r = wm * 64 + m * 16 + l16;
            #pragma unroll
            for (int j = 0; j < 16; ++j) {
                const int ko = j + ((j >= 8) ? 8 : 0) + half * 8;
                u.s[j] = As[r * BK + ko];
            }
            afr[m] = __builtin_bit_cast(v16bf, u.v);
        }
        #pragma unroll
        for (int n = 0; n < 2; ++n) {
            union { v16u v; unsigned short s[16]; } u;
            const int c = wn * 32 + n * 16 + l16;
            #pragma unroll
            for (int j = 0; j < 16; ++j) {
                const int ko = j + ((j >= 8) ? 8 : 0) + half * 8;
                u.s[j] = Bs[c][ko];
            }
            bfr[n] = __builtin_bit_cast(v16bf, u.v);
        }

        #pragma unroll
        for (int m = 0; m < 4; ++m)
            #pragma unroll
            for (int n = 0; n < 2; ++n)
                acc[m][n] = __builtin_amdgcn_wmma_f32_16x16x32_bf16(
                    false, afr[m], false, bfr[n], (short)0, acc[m][n], false, false);

        __syncthreads();
    }

    // ---- epilogue + store. C/D layout: N = l16 ; row = half*8 + r
    #pragma unroll
    for (int m = 0; m < 4; ++m) {
        #pragma unroll
        for (int n = 0; n < 2; ++n) {
            const int col = colBase + wn * 32 + n * 16 + l16;
            if (col >= N) continue;
            const int row0 = rowBase + wm * 64 + m * 16 + half * 8;
            #pragma unroll
            for (int r = 0; r < 8; ++r) {
                const size_t idx = (size_t)(row0 + r) * N + col;
                float v = acc[m][n][r];
                if (mode == 1)      v = v + aux[idx];
                else if (mode == 2) v = sigmoidf_(v);
                else if (mode == 3) v = siluf_(v) * aux[idx];
                if (obf) ((unsigned short*)Cv)[idx] = f2bf(v);
                else     ((float*)Cv)[idx] = v;
            }
        }
    }
}

// ---------------------------------------------------------------------------
// RMSNorm over last dim; output bf16 (feeds GEMM A operands only).
// ---------------------------------------------------------------------------
__global__ __launch_bounds__(256) void rmsnorm_kernel(
    const float* __restrict__ x, const float* __restrict__ w,
    unsigned short* __restrict__ out, int cols)
{
    __shared__ float red[256];
    const size_t row = blockIdx.x;
    const float* xr = x + row * cols;
    float ss = 0.f;
    for (int c = threadIdx.x; c < cols; c += 256) { float v = xr[c]; ss += v * v; }
    red[threadIdx.x] = ss; __syncthreads();
    for (int s = 128; s > 0; s >>= 1) {
        if (threadIdx.x < (unsigned)s) red[threadIdx.x] += red[threadIdx.x + s];
        __syncthreads();
    }
    const float rs = rsqrtf(red[0] / (float)cols + EPS);
    for (int c = threadIdx.x; c < cols; c += 256)
        out[row * cols + c] = f2bf(xr[c] * rs * w[c]);
}

// ---------------------------------------------------------------------------
// Depthwise causal conv1d (K=4) + SiLU over [B,S,P] (flattened [M,P]).
// ---------------------------------------------------------------------------
__global__ __launch_bounds__(256) void conv_silu_kernel(
    const float* __restrict__ pre, const float* __restrict__ w,
    float* __restrict__ post)
{
    const size_t gid = (size_t)blockIdx.x * 256 + threadIdx.x;  // < M*P
    const int    p   = (int)(gid % PP);
    const size_t row = gid / PP;
    const int    s   = (int)(row % SS);
    float acc = 0.f;
    #pragma unroll
    for (int j = 0; j < KCONV; ++j) {
        const int sj = s + j - (KCONV - 1);
        if (sj >= 0)
            acc += pre[(row + j - (KCONV - 1)) * PP + p] * w[p * KCONV + j];
    }
    post[gid] = siluf_(acc);
}

// ---------------------------------------------------------------------------
// KDA gate prep (in place): g = -exp(A_log[h]) * softplus(g + dt_bias[p])
// ---------------------------------------------------------------------------
__global__ __launch_bounds__(256) void kda_prep_kernel(
    float* __restrict__ g, const float* __restrict__ dt_bias,
    const float* __restrict__ A_log)
{
    const size_t gid = (size_t)blockIdx.x * 256 + threadIdx.x;  // < M*P
    const int p = (int)(gid % PP);
    const int h = p / HDIM;
    const float xg = g[gid] + dt_bias[p];
    const float sp = (xg > 20.f) ? xg : log1pf(__expf(xg));
    g[gid] = -__expf(A_log[h]) * sp;
}

// ---------------------------------------------------------------------------
// L2-normalize rows of HDIM=128 in place, then scale.
// ---------------------------------------------------------------------------
__global__ __launch_bounds__(128) void l2norm_kernel(float* __restrict__ x, float scale)
{
    __shared__ float red[128];
    const size_t row = blockIdx.x;                  // < M*H
    float* xr = x + row * HDIM;
    const float v = xr[threadIdx.x];
    red[threadIdx.x] = v * v; __syncthreads();
    for (int s = 64; s > 0; s >>= 1) {
        if (threadIdx.x < (unsigned)s) red[threadIdx.x] += red[threadIdx.x + s];
        __syncthreads();
    }
    const float rs = rsqrtf(red[0] + 1e-6f);
    xr[threadIdx.x] = v * rs * scale;
}

// ---------------------------------------------------------------------------
// KDA delta-rule scan. Grid = B*H, 128 threads. Lane v owns state column
// S[:,v] in 128 f32 registers; per-step vectors staged in LDS.
// ---------------------------------------------------------------------------
__global__ __launch_bounds__(128) void kda_kernel(
    const float* __restrict__ q, const float* __restrict__ k,
    const float* __restrict__ v, const float* __restrict__ g,
    const float* __restrict__ beta, float* __restrict__ o)
{
    const int bh  = blockIdx.x;           // 0..63
    const int b   = bh / HH;
    const int h   = bh % HH;
    const int tid = threadIdx.x;          // value channel

    __shared__ float sq[HDIM], sk[HDIM], se[HDIM];
    __shared__ float sbeta;

    float Scol[HDIM];
    #pragma unroll
    for (int i = 0; i < HDIM; ++i) Scol[i] = 0.f;

    for (int t = 0; t < SS; ++t) {
        const size_t base = ((size_t)(b * SS + t)) * PP + (size_t)h * HDIM;
        sq[tid] = q[base + tid];
        sk[tid] = k[base + tid];
        se[tid] = __expf(g[base + tid]);
        if (tid == 0) sbeta = beta[(size_t)(b * SS + t) * HH + h];
        __syncthreads();

        const float vt = v[base + tid];
        float err = 0.f;
        #pragma unroll
        for (int kk = 0; kk < HDIM; ++kk) {
            Scol[kk] *= se[kk];
            err = fmaf(sk[kk], Scol[kk], err);
        }
        const float delta = sbeta * (vt - err);
        float ot = 0.f;
        #pragma unroll
        for (int kk = 0; kk < HDIM; ++kk) {
            Scol[kk] = fmaf(sk[kk], delta, Scol[kk]);
            ot = fmaf(sq[kk], Scol[kk], ot);
        }
        o[base + tid] = ot;
        __syncthreads();
    }
}

// ---------------------------------------------------------------------------
// Gated RMSNorm -> bf16 (feeds wo GEMM as A operand).
// ---------------------------------------------------------------------------
__global__ __launch_bounds__(128) void gated_norm_kernel(
    const float* __restrict__ o, const float* __restrict__ gate,
    const float* __restrict__ w, unsigned short* __restrict__ out)
{
    __shared__ float red[128];
    const size_t row = blockIdx.x;                  // < M*H
    const size_t idx = row * HDIM + threadIdx.x;
    const float og = o[idx] * sigmoidf_(gate[idx]);
    red[threadIdx.x] = og * og; __syncthreads();
    for (int s = 64; s > 0; s >>= 1) {
        if (threadIdx.x < (unsigned)s) red[threadIdx.x] += red[threadIdx.x + s];
        __syncthreads();
    }
    const float rs = rsqrtf(red[0] / (float)HDIM + EPS);
    out[idx] = f2bf(og * rs * w[threadIdx.x]);
}

// ---------------------------------------------------------------------------
// Host-side orchestration
// ---------------------------------------------------------------------------
extern "C" void kernel_launch(void* const* d_in, const int* in_sizes, int n_in,
                              void* d_out, int out_size, void* d_ws, size_t ws_size,
                              hipStream_t stream)
{
    const float* x       = (const float*)d_in[0];
    const float* norm1_w = (const float*)d_in[1];
    const float* wq      = (const float*)d_in[2];
    const float* wk      = (const float*)d_in[3];
    const float* wv      = (const float*)d_in[4];
    const float* conv_q  = (const float*)d_in[5];
    const float* conv_k  = (const float*)d_in[6];
    const float* conv_v  = (const float*)d_in[7];
    const float* f_a     = (const float*)d_in[8];
    const float* f_b     = (const float*)d_in[9];
    const float* A_log   = (const float*)d_in[10];
    const float* dt_bias = (const float*)d_in[11];
    const float* b_w     = (const float*)d_in[12];
    const float* g_a     = (const float*)d_in[13];
    const float* g_b     = (const float*)d_in[14];
    const float* onorm_w = (const float*)d_in[15];
    const float* wo      = (const float*)d_in[16];
    const float* norm2_w = (const float*)d_in[17];
    const float* w1      = (const float*)d_in[18];
    const float* w2      = (const float*)d_in[19];
    const float* w3      = (const float*)d_in[20];
    float* out = (float*)d_out;
    float* ws  = (float*)d_ws;

    // Workspace arena (f32 element units). Aliased reuse:
    //   G->QPRE, GATE->KPRE, O->VPRE, ONORM_BF->QPOST,
    //   FF3(f32)->QPRE..KPRE (contiguous 2*MP), FF1_BF(ushort)->VPRE.
    constexpr size_t MD  = (size_t)MM * DDIM;
    constexpr size_t MP  = (size_t)MM * PP;
    constexpr size_t O_HBF    = 0;                          // ushort[MD]
    constexpr size_t O_QPRE   = O_HBF    + MD / 2;
    constexpr size_t O_KPRE   = O_QPRE   + MP;
    constexpr size_t O_VPRE   = O_KPRE   + MP;
    constexpr size_t O_QPOST  = O_VPRE   + MP;
    constexpr size_t O_KPOST  = O_QPOST  + MP;
    constexpr size_t O_VPOST  = O_KPOST  + MP;
    constexpr size_t O_THDBF  = O_VPOST  + MP;              // ushort[M*HDIM]
    constexpr size_t O_BETA   = O_THDBF  + ((size_t)MM * HDIM) / 2;
    constexpr size_t O_X2     = O_BETA   + (size_t)MM * HH;
    constexpr size_t O_H2BF   = O_X2     + MD;              // ushort[MD]
    constexpr size_t O_WB     = O_H2BF   + MD / 2;          // bf16 weight arena

    unsigned short* HBF   = (unsigned short*)(ws + O_HBF);
    unsigned short* THDBF = (unsigned short*)(ws + O_THDBF);
    unsigned short* H2BF  = (unsigned short*)(ws + O_H2BF);
    unsigned short* ONBF  = (unsigned short*)(ws + O_QPOST);
    unsigned short* FF1BF = (unsigned short*)(ws + O_VPRE);
    unsigned short* wb    = (unsigned short*)(ws + O_WB);

    size_t woff = 0;
    auto cvt = [&](const float* src, size_t n) -> const unsigned short* {
        unsigned short* dst = wb + woff;
        woff += n;
        f32_to_bf16_kernel<<<(int)(n / 1024), 256, 0, stream>>>(src, dst, (int)(n / 4));
        return dst;
    };

    // Convert all GEMM weights to bf16 once per launch.
    const unsigned short* WQB = cvt(wq,  (size_t)DDIM * PP);
    const unsigned short* WKB = cvt(wk,  (size_t)DDIM * PP);
    const unsigned short* WVB = cvt(wv,  (size_t)DDIM * PP);
    const unsigned short* FAB = cvt(f_a, (size_t)DDIM * HDIM);
    const unsigned short* FBB = cvt(f_b, (size_t)HDIM * PP);
    const unsigned short* BWB = cvt(b_w, (size_t)DDIM * HH);
    const unsigned short* GAB = cvt(g_a, (size_t)DDIM * HDIM);
    const unsigned short* GBB = cvt(g_b, (size_t)HDIM * PP);
    const unsigned short* WOB = cvt(wo,  (size_t)PP * DDIM);
    const unsigned short* W1B = cvt(w1,  (size_t)DDIM * FDIM);
    const unsigned short* W2B = cvt(w2,  (size_t)FDIM * DDIM);
    const unsigned short* W3B = cvt(w3,  (size_t)DDIM * FDIM);

    auto gemm = [&](const unsigned short* A, const unsigned short* Bm, void* C,
                    const float* aux, int M_, int N_, int K_, int mode, int obf) {
        dim3 grid((N_ + BN - 1) / BN, (M_ + BM - 1) / BM);
        gemm_bf16_wmma<<<grid, 256, 0, stream>>>(A, Bm, C, aux, M_, N_, K_, mode, obf);
    };

    const int ELT_BLKS = (int)(((size_t)MM * PP) / 256);
    const float qscale = 0.08838834764831845f;   // 1/sqrt(128)

    // 1) h = rmsnorm(x) -> bf16
    rmsnorm_kernel<<<MM, 256, 0, stream>>>(x, norm1_w, HBF, DDIM);
    // 2) q,k,v pre-conv projections (f32 out)
    gemm(HBF, WQB, ws + O_QPRE, nullptr, MM, PP, DDIM, 0, 0);
    gemm(HBF, WKB, ws + O_KPRE, nullptr, MM, PP, DDIM, 0, 0);
    gemm(HBF, WVB, ws + O_VPRE, nullptr, MM, PP, DDIM, 0, 0);
    // 3) depthwise causal conv + silu
    conv_silu_kernel<<<ELT_BLKS, 256, 0, stream>>>(ws + O_QPRE, conv_q, ws + O_QPOST);
    conv_silu_kernel<<<ELT_BLKS, 256, 0, stream>>>(ws + O_KPRE, conv_k, ws + O_KPOST);
    conv_silu_kernel<<<ELT_BLKS, 256, 0, stream>>>(ws + O_VPRE, conv_v, ws + O_VPOST);
    // 4) KDA decay gate: g = -exp(A_log)*softplus((h@f_a)@f_b + dt_bias)
    gemm(HBF, FAB, THDBF, nullptr, MM, HDIM, DDIM, 0, 1);
    gemm(THDBF, FBB, ws + O_QPRE /*G*/, nullptr, MM, PP, HDIM, 0, 0);
    kda_prep_kernel<<<ELT_BLKS, 256, 0, stream>>>(ws + O_QPRE, dt_bias, A_log);
    // 5) beta = sigmoid(h@b_w)
    gemm(HBF, BWB, ws + O_BETA, nullptr, MM, HH, DDIM, 2, 0);
    // 6) l2norm(q)*scale, l2norm(k)
    l2norm_kernel<<<MM * HH, 128, 0, stream>>>(ws + O_QPOST, qscale);
    l2norm_kernel<<<MM * HH, 128, 0, stream>>>(ws + O_KPOST, 1.0f);
    // 7) output gate (pre-sigmoid): (h@g_a)@g_b
    gemm(HBF, GAB, THDBF, nullptr, MM, HDIM, DDIM, 0, 1);
    gemm(THDBF, GBB, ws + O_KPRE /*GATE*/, nullptr, MM, PP, HDIM, 0, 0);
    // 8) KDA scan -> O (f32 at VPRE)
    kda_kernel<<<BB * HH, 128, 0, stream>>>(ws + O_QPOST, ws + O_KPOST, ws + O_VPOST,
                                            ws + O_QPRE, ws + O_BETA, ws + O_VPRE);
    // 9) gated RMSNorm -> ONORM bf16 (over q region, q already consumed)
    gated_norm_kernel<<<MM * HH, 128, 0, stream>>>(ws + O_VPRE, ws + O_KPRE, onorm_w, ONBF);
    // 10) x2 = x + onorm@wo
    gemm(ONBF, WOB, ws + O_X2, x, MM, DDIM, PP, 1, 0);
    // 11) h2 = rmsnorm(x2) -> bf16
    rmsnorm_kernel<<<MM, 256, 0, stream>>>(ws + O_X2, norm2_w, H2BF, DDIM);
    // 12) FF3 = h2@w3 (f32) ; FF1 = silu(h2@w1)*FF3 -> bf16
    gemm(H2BF, W3B, ws + O_QPRE /*FF3*/, nullptr, MM, FDIM, DDIM, 0, 0);
    gemm(H2BF, W1B, FF1BF, ws + O_QPRE, MM, FDIM, DDIM, 3, 1);
    // 13) out = x2 + FF1@w2
    gemm(FF1BF, W2B, out, ws + O_X2, MM, DDIM, FDIM, 1, 0);

    (void)in_sizes; (void)n_in; (void)out_size; (void)ws_size;
}
